// MultiHeadAttention_66958540144924
// MI455X (gfx1250) — matrix operand order repaired and checked by reading, not compiled, not run on hardware
//
#include <hip/hip_runtime.h>
#include <hip/hip_bf16.h>

typedef _Float16 v8h  __attribute__((ext_vector_type(8)));
typedef _Float16 v16h __attribute__((ext_vector_type(16)));
typedef float    v8f  __attribute__((ext_vector_type(8)));

#define DIMC   1024
#define NHEADS 16
#define HD     64
#define BATCH  2
#define SEQ    2048
#define ROWS   (BATCH*SEQ)   // 4096

// ---------------- helpers ----------------

__device__ __forceinline__ v16h cat8(v8h lo, v8h hi) {
  return __builtin_shufflevector(lo, hi, 0,1,2,3,4,5,6,7,8,9,10,11,12,13,14,15);
}

__device__ __forceinline__ v8f wmma_f16(v16h a, v16h b, v8f c) {
  return __builtin_amdgcn_wmma_f32_16x16x32_f16(false, a, false, b, (short)0, c, false, false);
}

// ds_swizzle xor within groups of 32; masks < 16 keep lanes inside their 16-lane half
template<int MASK>
__device__ __forceinline__ float swz_xor(float x) {
  return __int_as_float(__builtin_amdgcn_ds_swizzle(__float_as_int(x), (MASK<<10) | 0x1f));
}
__device__ __forceinline__ float rowmax16(float x){
  x = fmaxf(x, swz_xor<1>(x));
  x = fmaxf(x, swz_xor<2>(x));
  x = fmaxf(x, swz_xor<4>(x));
  x = fmaxf(x, swz_xor<8>(x));
  return x;
}
__device__ __forceinline__ float rowsum16(float x){
  x += swz_xor<1>(x);
  x += swz_xor<2>(x);
  x += swz_xor<4>(x);
  x += swz_xor<8>(x);
  return x;
}

// ---------------- f32 -> f16 convert ----------------

__global__ void f32_to_f16(const float* __restrict__ s, _Float16* __restrict__ d, int n){
  for (int i = blockIdx.x*blockDim.x + threadIdx.x; i < n; i += gridDim.x*blockDim.x)
    d[i] = (_Float16)s[i];
}

// ---------------- WMMA GEMM: C[M,Nout] = X[M,K] * W[Nout,K]^T ----------------
// one wave -> 32x64 strip of C (8 accumulators), register double-buffered over K
// so the next k-step's fragments load while the current 8 WMMAs execute.

__global__ void __launch_bounds__(256) gemm_xwt(const _Float16* __restrict__ X,
                                                const _Float16* __restrict__ W,
                                                float* __restrict__ C,
                                                int K, int Nout){
  const int lane = threadIdx.x & 31;
  const int wave = threadIdx.x >> 5;
  const int lo16 = lane & 15;
  const int half = lane >> 4;
  const int tm   = blockIdx.y*8 + wave;   // 32-row tile index
  const int nb   = blockIdx.x*64;         // 64-col tile base

  const _Float16* ar0 = X + (size_t)(tm*32 +      lo16) * K;
  const _Float16* ar1 = X + (size_t)(tm*32 + 16 + lo16) * K;
  const _Float16* wr0 = W + (size_t)(nb +  0 + lo16) * K;
  const _Float16* wr1 = W + (size_t)(nb + 16 + lo16) * K;
  const _Float16* wr2 = W + (size_t)(nb + 32 + lo16) * K;
  const _Float16* wr3 = W + (size_t)(nb + 48 + lo16) * K;

  v8f c00 = {}, c01 = {}, c02 = {}, c03 = {};
  v8f c10 = {}, c11 = {}, c12 = {}, c13 = {};

  // current fragments (k = 0)
  v16h a0 = cat8(*(const v8h*)(ar0 + half*16), *(const v8h*)(ar0 + 8 + half*16));
  v16h a1 = cat8(*(const v8h*)(ar1 + half*16), *(const v8h*)(ar1 + 8 + half*16));
  v16h b0 = *(const v16h*)(wr0 + half*16);
  v16h b1 = *(const v16h*)(wr1 + half*16);
  v16h b2 = *(const v16h*)(wr2 + half*16);
  v16h b3 = *(const v16h*)(wr3 + half*16);

  for (int k0 = 0; k0 < K - 32; k0 += 32){
    const int kn = k0 + 32;
    // issue next k-step's loads; they fly during the 8 WMMAs below
    v16h na0 = cat8(*(const v8h*)(ar0 + kn + half*16), *(const v8h*)(ar0 + kn + 8 + half*16));
    v16h na1 = cat8(*(const v8h*)(ar1 + kn + half*16), *(const v8h*)(ar1 + kn + 8 + half*16));
    v16h nb0 = *(const v16h*)(wr0 + kn + half*16);
    v16h nb1 = *(const v16h*)(wr1 + kn + half*16);
    v16h nb2 = *(const v16h*)(wr2 + kn + half*16);
    v16h nb3 = *(const v16h*)(wr3 + kn + half*16);

    c00 = wmma_f16(a0, b0, c00);
    c01 = wmma_f16(a0, b1, c01);
    c02 = wmma_f16(a0, b2, c02);
    c03 = wmma_f16(a0, b3, c03);
    c10 = wmma_f16(a1, b0, c10);
    c11 = wmma_f16(a1, b1, c11);
    c12 = wmma_f16(a1, b2, c12);
    c13 = wmma_f16(a1, b3, c13);

    a0 = na0; a1 = na1; b0 = nb0; b1 = nb1; b2 = nb2; b3 = nb3;
  }
  // tail k-step
  c00 = wmma_f16(a0, b0, c00);
  c01 = wmma_f16(a0, b1, c01);
  c02 = wmma_f16(a0, b2, c02);
  c03 = wmma_f16(a0, b3, c03);
  c10 = wmma_f16(a1, b0, c10);
  c11 = wmma_f16(a1, b1, c11);
  c12 = wmma_f16(a1, b2, c12);
  c13 = wmma_f16(a1, b3, c13);

  #pragma unroll
  for (int r = 0; r < 8; ++r){
    int m0 = tm*32 + r + 8*half;          // C-layout: M = r + 8*(lane/16)
    float* row0 = C + (size_t)m0*Nout + nb + lo16;
    row0[ 0] = c00[r];
    row0[16] = c01[r];
    row0[32] = c02[r];
    row0[48] = c03[r];
    float* row1 = C + (size_t)(m0 + 16)*Nout + nb + lo16;
    row1[ 0] = c10[r];
    row1[16] = c11[r];
    row1[32] = c12[r];
    row1[48] = c13[r];
  }
}

// ---------------- RoPE + pack (q scaled by 1/sqrt(hd)) ----------------
// qkv f32 [B,N,3,H,hd]  ->  Q/K/V f16 [B,H,N,hd]

__global__ void rope_pack(const float* __restrict__ qkv,
                          _Float16* __restrict__ Q,
                          _Float16* __restrict__ Kp,
                          _Float16* __restrict__ V){
  int idx = blockIdx.x*blockDim.x + threadIdx.x;   // ((b*16+h)*2048+n)*64+d
  if (idx >= BATCH*NHEADS*SEQ*HD) return;
  int d = idx & 63;
  int n = (idx >> 6) & 2047;
  int h = (idx >> 17) & 15;
  int b = idx >> 21;

  size_t row  = (size_t)(b*SEQ + n) * (3*DIMC);
  int col  = h*HD + d;
  int colr = h*HD + ((d < 32) ? d + 32 : d - 32);

  float q  = qkv[row +          col];
  float qo = qkv[row +          colr];
  float k  = qkv[row + DIMC   + col];
  float ko = qkv[row + DIMC   + colr];
  float v  = qkv[row + 2*DIMC + col];

  int j = d & 31;
  float inv = exp2f(-((float)(2*j) / (float)HD) * 13.287712379549449f); // log2(10000)
  float ang = (float)n * inv;
  float c = cosf(ang), s = sinf(ang);
  float qr = (q*c + ((d < 32) ? -qo : qo)*s) * 0.125f;   // fold 1/sqrt(64)
  float kr =  k*c + ((d < 32) ? -ko : ko)*s;

  Q[idx]  = (_Float16)qr;
  Kp[idx] = (_Float16)kr;
  V[idx]  = (_Float16)v;
}

// ---------------- flash attention ----------------
// block = 8 waves; each wave owns 16 q rows. Double-buffered 32-row K/V tiles in
// LDS, one barrier per tile; next tile's global loads overlap current compute.

__global__ void __launch_bounds__(256) flash_attn(const _Float16* __restrict__ Q,
                                                  const _Float16* __restrict__ Kp,
                                                  const _Float16* __restrict__ V,
                                                  _Float16* __restrict__ O){
  __shared__ alignas(32) _Float16 Kl[2][32*64];   // K tile row-major [kv][hd]
  __shared__ alignas(32) _Float16 Vt[2][64*32];   // V tile transposed [hd][kv]
  __shared__ alignas(32) _Float16 Pl[8][16*32];   // per-wave P staging (C->A layout)

  const int lane = threadIdx.x & 31;
  const int wave = threadIdx.x >> 5;
  const int lo16 = lane & 15;
  const int half = lane >> 4;
  const int bh   = blockIdx.y;                    // b*16 + h
  const int qrow0 = blockIdx.x*128 + wave*16;
  const size_t base = (size_t)bh * SEQ * HD;

  // q fragments for this wave's 16 rows (hd=64 -> two K=32 fragments)
  const _Float16* qrow = Q + base + (size_t)(qrow0 + lo16) * HD;
  v16h qa0 = cat8(*(const v8h*)(qrow +  0 + half*16), *(const v8h*)(qrow +  8 + half*16));
  v16h qa1 = cat8(*(const v8h*)(qrow + 32 + half*16), *(const v8h*)(qrow + 40 + half*16));

  v8f a0 = {}, a1 = {}, a2 = {}, a3 = {};
  float mrow[8], lrow[8];
  #pragma unroll
  for (int r = 0; r < 8; ++r){ mrow[r] = -1e30f; lrow[r] = 0.f; }

  const int tid = threadIdx.x;
  const int kr  = (tid*8) >> 6;                   // kv row this thread stages
  const int cc  = (tid*8) & 63;                   // hd col base (8 halfs)
  const _Float16* ksrc = Kp + base + (size_t)kr*HD + cc;
  const _Float16* vsrc = V  + base + (size_t)kr*HD + cc;

  // preload tile 0 into registers
  v8h kk = *(const v8h*)(ksrc);
  v8h vv = *(const v8h*)(vsrc);

  const int NT = SEQ/32;                          // 64 kv tiles
  for (int i = 0; i < NT; ++i){
    const int buf = i & 1;
    // ---- stage current tile registers into LDS ----
    *(v8h*)(&Kl[buf][kr*64 + cc]) = kk;
    #pragma unroll
    for (int j = 0; j < 8; ++j) Vt[buf][(cc + j)*32 + kr] = vv[j];

    // ---- issue next tile's global loads (overlap with compute below) ----
    v8h nk = kk, nv = vv;
    if (i + 1 < NT){
      nk = *(const v8h*)(ksrc + (size_t)(i+1)*32*HD);
      nv = *(const v8h*)(vsrc + (size_t)(i+1)*32*HD);
    }

    __syncthreads();   // staging of buf visible; prior tile's reads complete

    // ---- scores: 16 x 32 (two 16x16 tiles, K=hd=64 -> 2 WMMA each) ----
    v8f s0 = {}, s1 = {};
    {
      v16h b00 = *(const v16h*)(&Kl[buf][(     lo16)*64 +  0 + half*16]);
      v16h b01 = *(const v16h*)(&Kl[buf][(     lo16)*64 + 32 + half*16]);
      v16h b10 = *(const v16h*)(&Kl[buf][(16 + lo16)*64 +  0 + half*16]);
      v16h b11 = *(const v16h*)(&Kl[buf][(16 + lo16)*64 + 32 + half*16]);
      s0 = wmma_f16(qa0, b00, s0);
      s0 = wmma_f16(qa1, b01, s0);
      s1 = wmma_f16(qa0, b10, s1);
      s1 = wmma_f16(qa1, b11, s1);
    }

    // ---- online softmax (row = r + 8*half across 16 lanes of this half) ----
    #pragma unroll
    for (int r = 0; r < 8; ++r){
      float v0 = s0[r], v1 = s1[r];
      float tmax = rowmax16(fmaxf(v0, v1));
      float nm = fmaxf(mrow[r], tmax);
      float cf = __expf(mrow[r] - nm);
      float p0 = __expf(v0 - nm);
      float p1 = __expf(v1 - nm);
      lrow[r] = lrow[r]*cf + p0 + p1;
      mrow[r] = nm;
      a0[r] *= cf; a1[r] *= cf; a2[r] *= cf; a3[r] *= cf;
      int mr = r + 8*half;
      Pl[wave][mr*32 +      lo16] = (_Float16)p0;
      Pl[wave][mr*32 + 16 + lo16] = (_Float16)p1;
    }
    __builtin_amdgcn_wave_barrier();   // per-wave LDS is in-order; keep sched order
    asm volatile("" ::: "memory");

    // ---- P (A-layout) x V_T (B-layout) -> out accumulators ----
    v16h pa = cat8(*(const v8h*)(&Pl[wave][lo16*32 +     half*16]),
                   *(const v8h*)(&Pl[wave][lo16*32 + 8 + half*16]));
    v16h vb0 = *(const v16h*)(&Vt[buf][( 0 + lo16)*32 + half*16]);
    v16h vb1 = *(const v16h*)(&Vt[buf][(16 + lo16)*32 + half*16]);
    v16h vb2 = *(const v16h*)(&Vt[buf][(32 + lo16)*32 + half*16]);
    v16h vb3 = *(const v16h*)(&Vt[buf][(48 + lo16)*32 + half*16]);
    a0 = wmma_f16(pa, vb0, a0);
    a1 = wmma_f16(pa, vb1, a1);
    a2 = wmma_f16(pa, vb2, a2);
    a3 = wmma_f16(pa, vb3, a3);

    kk = nk; vv = nv;
  }

  // ---- epilogue: normalize and store as [b, n, h*64+d] f16 ----
  const int b = bh >> 4, h = bh & 15;
  #pragma unroll
  for (int r = 0; r < 8; ++r){
    float L = rowsum16(lrow[r]);
    float inv = 1.0f / L;
    int mr = qrow0 + r + 8*half;
    _Float16* orow = O + ((size_t)(b*SEQ + mr))*DIMC + h*HD;
    orow[ 0 + lo16] = (_Float16)(a0[r]*inv);
    orow[16 + lo16] = (_Float16)(a1[r]*inv);
    orow[32 + lo16] = (_Float16)(a2[r]*inv);
    orow[48 + lo16] = (_Float16)(a3[r]*inv);
  }
}

// ---------------- launch ----------------
// workspace layout (bytes, all 256-aligned); total ~101 MB (< 192 MB L2)

extern "C" void kernel_launch(void* const* d_in, const int* in_sizes, int n_in,
                              void* d_out, int out_size, void* d_ws, size_t ws_size,
                              hipStream_t stream) {
  (void)in_sizes; (void)n_in; (void)out_size; (void)ws_size;
  const float* x    = (const float*)d_in[0];
  const float* wqkv = (const float*)d_in[1];
  const float* wout = (const float*)d_in[2];

  char* ws = (char*)d_ws;
  _Float16* XH   = (_Float16*)(ws +         0);   //  8.0 MB  x f16
  _Float16* WQH  = (_Float16*)(ws +  8388608);    //  6.0 MB  w_qkv f16
  _Float16* WOH  = (_Float16*)(ws + 14680064);    //  2.0 MB  w_out f16
  float*    QKVF = (float*)   (ws + 16777216);    // 48.0 MB  qkv f32
  _Float16* QH   = (_Float16*)(ws + 67108864);    //  8.0 MB  q f16 [B,H,N,hd]
  _Float16* KH   = (_Float16*)(ws + 75497472);    //  8.0 MB  k f16
  _Float16* VH   = (_Float16*)(ws + 83886080);    //  8.0 MB  v f16
  _Float16* AOH  = (_Float16*)(ws + 92274688);    //  8.0 MB  attn out f16 [B,N,C]

  f32_to_f16<<<2048, 256, 0, stream>>>(x,    XH,  ROWS*DIMC);
  f32_to_f16<<<2048, 256, 0, stream>>>(wqkv, WQH, 3*DIMC*DIMC);
  f32_to_f16<<<1024, 256, 0, stream>>>(wout, WOH, DIMC*DIMC);

  // qkv = x @ w_qkv^T : [4096,1024] x [3072,1024]^T   (M tiles of 32 rows)
  gemm_xwt<<<dim3(48, 16), 256, 0, stream>>>(XH, WQH, QKVF, DIMC, 3*DIMC);

  rope_pack<<<(BATCH*NHEADS*SEQ*HD)/256, 256, 0, stream>>>(QKVF, QH, KH, VH);

  flash_attn<<<dim3(SEQ/128, BATCH*NHEADS), 256, 0, stream>>>(QH, KH, VH, AOH);

  // out = attn_out @ w_out^T : [4096,1024] x [1024,1024]^T
  gemm_xwt<<<dim3(16, 16), 256, 0, stream>>>(AOH, WOH, (float*)d_out, DIMC, DIMC);
}
